// classSR_3class_fused_fsrcnn_net_3865470566901
// MI455X (gfx1250) — matrix operand order repaired and checked by reading, not compiled, not run on hardware
//
#include <hip/hip_runtime.h>
#include <hip/hip_bf16.h>

// ============================================================================
// ClassSR (3-class router -> 3 FSRCNN experts) for MI455X / gfx1250.
//
//  * Output write (403 MB f32 @ 23.3 TB/s ~= 17us) is the traffic floor; the
//    ConvTranspose tail is fused with the scatter so it is written once.
//  * Every conv = im2col GEMM on v_wmma_f32_16x16x32_f16 tiles, f16 data /
//    f32 accum. One block per image: weights (f16, K-padded) + full input
//    feature map staged in LDS (<=114 KB << 320 KB/WGP). A fragments = two
//    ds_load_b128; B fragments = branch-free ds gathers. 4-way N register
//    blocking: one A fragment feeds 4 WMMAs.
//  * ALL geometry is compile-time (<KH,KW,STRIDE,PAD,LOGW,CIN> templates,
//    K-loop fully unrolled) -> zero runtime div/mod in device code; B-element
//    (ci,ky,kx) decompositions are constant-folded per half-wave and selected
//    with a single v_cndmask.
// ============================================================================

typedef __attribute__((ext_vector_type(16))) _Float16 v16h;
typedef __attribute__((ext_vector_type(8)))  _Float16 v8h;
typedef __attribute__((ext_vector_type(8)))  float    v8f;

constexpr int ilog2c(int v) { return v <= 1 ? 0 : 1 + ilog2c(v >> 1); }

// ---------------------------------------------------------------------------
// utility kernels
// ---------------------------------------------------------------------------
__global__ void k_f32_to_f16(const float* __restrict__ src,
                             _Float16* __restrict__ dst, int n) {
  int i = blockIdx.x * blockDim.x + threadIdx.x;
  int stride = gridDim.x * blockDim.x;
  for (; i < n; i += stride) dst[i] = (_Float16)src[i];
}

__global__ void k_zero_f32(float* __restrict__ p, long n) {
  long i = (long)blockIdx.x * blockDim.x + threadIdx.x;
  long stride = (long)gridDim.x * blockDim.x;
  long n4 = n >> 2;
  float4* p4 = (float4*)p;
  for (long j = i; j < n4; j += stride) p4[j] = make_float4(0.f, 0.f, 0.f, 0.f);
  for (long j = (n4 << 2) + i; j < n; j += stride) p[j] = 0.f;
}

__global__ void k_fill_i32(int* __restrict__ p, int n, int v) {
  int i = blockIdx.x * blockDim.x + threadIdx.x;
  if (i < n) p[i] = v;
}

// ---------------------------------------------------------------------------
// LDS-staged im2col conv as WMMA GEMM.  One block (8 waves) per image n.
//   in  : f16 [N][CinAlloc][H*W]   (H = W = 1<<LOGW, square)
//   wgt : f32 [Cout][CIN][KH][KW]
//   out : f16 [N][CoutAlloc][Ho*Wo] (+bias, +PReLU(alpha)/LeakyReLU(leaky))
// ---------------------------------------------------------------------------
template <int KH, int KW, int STRIDE, int PAD, int LOGW, int CIN>
__global__ void __launch_bounds__(256)
k_conv_wmma(const _Float16* __restrict__ in, const float* __restrict__ wgt,
            const float* __restrict__ bias, const float* __restrict__ alpha,
            float leaky, _Float16* __restrict__ out,
            int CinAlloc, int Cout, int CoutAlloc) {
  extern __shared__ __align__(16) char smem_raw[];
  constexpr int KHW   = KH * KW;
  constexpr int W     = 1 << LOGW;
  constexpr int H     = W;
  constexpr int LOGHW = 2 * LOGW;
  constexpr int HW    = 1 << LOGHW;
  constexpr int Ho    = (H + 2 * PAD - KH) / STRIDE + 1;
  constexpr int Wo    = Ho;                 // square everywhere in this model
  constexpr int LOGWO = ilog2c(Wo);
  constexpr int P     = Ho * Wo;
  constexpr int K     = CIN * KHW;
  constexpr int KPAD  = (K + 31) & ~31;
  constexpr int NT    = (P + 15) / 16;
  constexpr int NG    = (NT + 3) / 4;       // groups of 4 pixel-tiles

  const int Mt    = (Cout + 15) >> 4;
  const int Mrows = Mt << 4;

  _Float16* sW  = (_Float16*)smem_raw;              // [Mrows][KPAD]
  _Float16* sIn = sW + (size_t)Mrows * KPAD;        // [CIN][H*W]

  const int n = blockIdx.x;

  // ---- stage weights f32->f16, zero-padded to [Mrows][KPAD] ---------------
  for (int i = threadIdx.x; i < Mrows * KPAD; i += blockDim.x) {
    int row = i / KPAD;                              // constexpr divisor
    int k   = i - row * KPAD;
    sW[i] = (row < Cout && k < K) ? (_Float16)wgt[row * K + k] : (_Float16)0.f;
  }
  // ---- stage this image's activations (contiguous, 16B vector copy) ------
  {
    const uint4* src = (const uint4*)(in + (size_t)n * CinAlloc * HW);
    uint4* dst = (uint4*)sIn;
    constexpr int N16 = (CIN * HW) / 8;              // halves / 8
    for (int i = threadIdx.x; i < N16; i += blockDim.x) dst[i] = src[i];
  }
  __syncthreads();

  const int lane = threadIdx.x & 31;
  const int hi   = lane >> 4;                        // ISA 7.12.2 half split
  const int lm   = lane & 15;
  const int wib  = threadIdx.x >> 5;
  const int wpb  = blockDim.x >> 5;

  const int tasks = Mt * NG;
  for (int t = wib; t < tasks; t += wpb) {
    const int mt = t / NG;                           // constexpr divisor
    const int ng = t - mt * NG;
    const int coA = (mt << 4) + lm;                  // A row of this lane
    const _Float16* arowBase = sW + coA * KPAD;

    int px[4], poy[4], pox[4];
    #pragma unroll
    for (int j = 0; j < 4; ++j) {
      px[j]  = (((ng << 2) + j) << 4) + lm;
      poy[j] = px[j] >> LOGWO;
      pox[j] = px[j] & (Wo - 1);
    }

    v8f c0 = {}, c1 = {}, c2 = {}, c3 = {};
    #pragma unroll
    for (int k0 = 0; k0 < KPAD; k0 += 32) {
      // A fragment: two aligned 16B LDS loads (elems 0-7 / 8-15).
      const _Float16* arow = arowBase + k0 + (hi << 3);
      v8h alo  = *(const v8h*)arow;
      v8h ahi8 = *(const v8h*)(arow + 16);
      v16h a;
      #pragma unroll
      for (int e = 0; e < 8; ++e) { a[e] = alo[e]; a[e + 8] = ahi8[e]; }

      // 4 B fragments (pixel tiles), each fed to one WMMA with the same A.
      #pragma unroll
      for (int j = 0; j < 4; ++j) {
        v16h b;
        #pragma unroll
        for (int e = 0; e < 16; ++e) {
          // both half-wave K indices are constants after full unroll:
          const int klo = k0 + (e < 8 ? e : 8 + e), khi = klo + 8;
          const int ciL = klo / KHW, rL = klo - ciL * KHW;
          const int ciH = khi / KHW, rH = khi - ciH * KHW;
          const int kyL = rL / KW, kxL = rL - kyL * KW;
          const int kyH = rH / KW, kxH = rH - kyH * KW;
          const bool vL = klo < K, vH = khi < K;
          int ci  = hi ? ciH : ciL;                  // one v_cndmask each
          int ky  = hi ? kyH : kyL;
          int kx  = hi ? kxH : kxL;
          bool kv = hi ? vH  : vL;
          int iy = poy[j] * STRIDE - PAD + ky;
          int ix = pox[j] * STRIDE - PAD + kx;
          bool inb = kv && (px[j] < P) &&
                     ((unsigned)iy < (unsigned)H) && ((unsigned)ix < (unsigned)W);
          int addr = (ci << LOGHW) + (iy << LOGW) + ix;
          addr = inb ? addr : 0;                     // branch-free clamped load
          _Float16 v = sIn[addr];
          b[e] = inb ? v : (_Float16)0.f;
        }
        v8f& cj = (j == 0) ? c0 : (j == 1) ? c1 : (j == 2) ? c2 : c3;
        cj = __builtin_amdgcn_wmma_f32_16x16x32_f16(
                 false, a, false, b, (short)0, cj, false, false);
      }
    }

    // ---- epilogue: bias + activation, f16 store ---------------------------
    #pragma unroll
    for (int j = 0; j < 4; ++j) {
      const int p = px[j];
      const v8f& cj = (j == 0) ? c0 : (j == 1) ? c1 : (j == 2) ? c2 : c3;
      #pragma unroll
      for (int r = 0; r < 8; ++r) {
        int oc = (mt << 4) + r + (hi << 3);
        if (oc < Cout && p < P) {
          float v = cj[r] + (bias ? bias[oc] : 0.f);
          if (alpha)            v = (v >= 0.f) ? v : alpha[oc] * v;  // PReLU
          else if (leaky > 0.f) v = (v >= 0.f) ? v : leaky * v;      // LReLU
          out[((long)n * CoutAlloc + oc) * P + p] = (_Float16)v;
        }
      }
    }
  }
}

// ---------------------------------------------------------------------------
// Classifier epilogue: global-avg-pool(32ch x 8x8) -> FC(32->3) -> argmax.
// ---------------------------------------------------------------------------
__global__ void k_route(const _Float16* __restrict__ h,  // [B][CAlloc][64]
                        const float* __restrict__ fcw,   // [3][32]
                        const float* __restrict__ fcb,   // [3]
                        int B, int CAlloc,
                        int* __restrict__ expert) {
  int img = blockIdx.x * blockDim.x + threadIdx.x;
  if (img >= B) return;
  const _Float16* hp = h + (long)img * CAlloc * 64;
  float pooled[32];
  #pragma unroll 4
  for (int c = 0; c < 32; ++c) {
    float s = 0.f;
    for (int q = 0; q < 64; ++q) s += (float)hp[c * 64 + q];
    pooled[c] = s * (1.f / 64.f);
  }
  float best = -3.4e38f; int be = 0;
  for (int j = 0; j < 3; ++j) {
    float l = fcb[j];
    for (int c = 0; c < 32; ++c) l += pooled[c] * fcw[j * 32 + c];
    if (l > best) { best = l; be = j; }   // first-max tie break == jnp.argmax
  }
  expert[img] = be;
}

// Serial capacity scan (rank = per-expert cumsum; overflow past cap dropped).
__global__ void k_scan(const int* __restrict__ expert, int B, int cap,
                       int* __restrict__ img_of_slot,   // [3*cap], prefilled -1
                       float* __restrict__ out_counts) { // 3 int32 (bit-cast)
  if (blockIdx.x != 0 || threadIdx.x != 0) return;
  int cnt[3] = {0, 0, 0};
  for (int i = 0; i < B; ++i) {
    int e = expert[i];
    int r = cnt[e]++;
    if (r < cap) img_of_slot[e * cap + r] = i;
  }
  int* oc = (int*)out_counts;
  for (int e = 0; e < 3; ++e) oc[e] = cnt[e];
}

// Gather dispatched images (f16) into this expert's capacity buffer.
__global__ void k_gather(const _Float16* __restrict__ xh,       // [B][3][1024]
                         const int* __restrict__ img_of_slot,   // [cap]
                         _Float16* __restrict__ feat, int CAlloc) {
  int slot = blockIdx.x / 12;
  int off  = (blockIdx.x % 12) * 256 + threadIdx.x;  // 0..3071 (3ch * 1024px)
  int img  = img_of_slot[slot];
  int c = off >> 10, pix = off & 1023;
  feat[((long)slot * CAlloc + c) * 1024 + pix] =
      (img >= 0) ? xh[((long)img * 3 + c) * 1024 + pix] : (_Float16)0.f;
}

// ---------------------------------------------------------------------------
// Fused ConvTranspose2d(d->3, k=9, s=4, pad=3, outpad=1) + scatter-combine.
// Writes the 403 MB output exactly once (memory-bound floor of the problem).
// ---------------------------------------------------------------------------
__global__ void k_tail_scatter(const _Float16* __restrict__ feat, // [cap][CAlloc][1024]
                               const float* __restrict__ tw,      // [d][3][9][9]
                               const float* __restrict__ tb,      // [3]
                               const int* __restrict__ img_of_slot,
                               float* __restrict__ out,           // [B][3][128*128]
                               int d, int CAlloc) {
  int slot = blockIdx.x >> 6;
  int pix  = ((blockIdx.x & 63) << 8) + threadIdx.x;  // 0..16383
  int img = img_of_slot[slot];
  if (img < 0) return;                                 // empty / overflow slot
  int oy = pix >> 7, ox = pix & 127;
  float a0 = tb[0], a1 = tb[1], a2 = tb[2];
  const _Float16* fs = feat + (long)slot * CAlloc * 1024;

  for (int ky = 0; ky < 9; ++ky) {
    int ty = oy + 3 - ky;
    if (ty < 0 || (ty & 3)) continue;
    int iy = ty >> 2;
    if (iy >= 32) continue;
    for (int kx = 0; kx < 9; ++kx) {
      int tx = ox + 3 - kx;
      if (tx < 0 || (tx & 3)) continue;
      int ix = tx >> 2;
      if (ix >= 32) continue;
      const _Float16* fp = fs + iy * 32 + ix;
      const float*    wp = tw + ky * 9 + kx;
      __builtin_prefetch(fp, 0, 3);   // global_prefetch_b8 on gfx1250
      for (int ci = 0; ci < d; ++ci) {
        float v = (float)fp[(long)ci * 1024];
        a0 += v * wp[(ci * 3 + 0) * 81];
        a1 += v * wp[(ci * 3 + 1) * 81];
        a2 += v * wp[(ci * 3 + 2) * 81];
      }
    }
  }
  long ob = (long)img * 3 * 16384 + pix;
  out[ob]             = a0;
  out[ob + 16384]     = a1;
  out[ob + 2 * 16384] = a2;
}

// ---------------------------------------------------------------------------
// host side
// ---------------------------------------------------------------------------
static inline size_t conv_shmem(int Cout, int Cin, int KH, int KW, int HW) {
  int K = Cin * KH * KW;
  int Kpad = (K + 31) & ~31;
  int Mrows = ((Cout + 15) >> 4) << 4;
  return (size_t)(Mrows * Kpad + Cin * HW) * 2;
}

// One expert pipeline; D is the FSRCNN feature width (compile-time).
template <int D>
static void run_expert(const float* const* prm, const int* sl,
                       const _Float16* xh, _Float16* featA, _Float16* featB,
                       float* out, hipStream_t stream) {
  constexpr int CAP = 768;
  dim3 blk(256);
  const float* head_w = prm[0];  const float* head_b = prm[1];
  const float* head_a = prm[2];  const float* shr_w  = prm[3];
  const float* shr_b  = prm[4];  const float* shr_a  = prm[5];
  const float* map_w  = prm[6];  const float* map_b  = prm[7];
  const float* map_a  = prm[8];  const float* exp_w  = prm[9];
  const float* exp_b  = prm[10]; const float* exp_a  = prm[11];
  const float* tail_w = prm[12]; const float* tail_b = prm[13];

  // gather dispatched images into capacity buffer
  k_gather<<<CAP * 12, blk, 0, stream>>>(xh, sl, featA, 64);
  // head: 5x5 conv 3->D, PReLU
  k_conv_wmma<5, 5, 1, 2, 5, 3>
      <<<CAP, blk, conv_shmem(D, 3, 5, 5, 1024), stream>>>(
          featA, head_w, head_b, head_a, 0.f, featB, 64, D, 64);
  // shrink: 1x1 D->12, PReLU
  k_conv_wmma<1, 1, 1, 0, 5, D>
      <<<CAP, blk, conv_shmem(12, D, 1, 1, 1024), stream>>>(
          featB, shr_w, shr_b, shr_a, 0.f, featA, 64, 12, 64);
  // 4 mapping layers: 3x3 12->12, PReLU (ping-pong)
  _Float16* mi = featA; _Float16* mo = featB;
  for (int m = 0; m < 4; ++m) {
    k_conv_wmma<3, 3, 1, 1, 5, 12>
        <<<CAP, blk, conv_shmem(12, 12, 3, 3, 1024), stream>>>(
            mi, map_w + (long)m * 12 * 12 * 9, map_b + m * 12, map_a + m * 12,
            0.f, mo, 64, 12, 64);
    _Float16* t = mi; mi = mo; mo = t;
  }
  // expand: 1x1 12->D, PReLU  (mi holds map3 output)
  k_conv_wmma<1, 1, 1, 0, 5, 12>
      <<<CAP, blk, conv_shmem(D, 12, 1, 1, 1024), stream>>>(
          mi, exp_w, exp_b, exp_a, 0.f, mo, 64, D, 64);
  // tail deconv fused with scatter-combine into d_out
  k_tail_scatter<<<CAP * 64, blk, 0, stream>>>(mo, tail_w, tail_b, sl, out,
                                               D, 64);
}

extern "C" void kernel_launch(void* const* d_in, const int* in_sizes, int n_in,
                              void* d_out, int out_size, void* d_ws, size_t ws_size,
                              hipStream_t stream) {
  (void)in_sizes; (void)n_in; (void)ws_size;
  constexpr int B = 2048, CAP = 768;

  // ---- inputs (setup_inputs dict order, nested dicts in insertion order) --
  const float* x = (const float*)d_in[0];
  const float* cw[5] = {(const float*)d_in[1], (const float*)d_in[3],
                        (const float*)d_in[5], (const float*)d_in[7],
                        (const float*)d_in[9]};
  const float* cb[5] = {(const float*)d_in[2], (const float*)d_in[4],
                        (const float*)d_in[6], (const float*)d_in[8],
                        (const float*)d_in[10]};
  const float* fcw = (const float*)d_in[11];
  const float* fcb = (const float*)d_in[12];

  // ---- workspace carve-out ------------------------------------------------
  char* ws = (char*)d_ws;
  size_t off = 0;
  auto carve = [&](size_t bytes) -> char* {
    size_t a = (off + 255) & ~(size_t)255;
    off = a + bytes;
    return ws + a;
  };
  _Float16* xh     = (_Float16*)carve((size_t)B * 3 * 1024 * 2);    // 12.6 MB
  _Float16* actA   = (_Float16*)carve((size_t)B * 128 * 64 * 2);    // 33.6 MB
  _Float16* actB   = (_Float16*)carve((size_t)B * 128 * 64 * 2);    // 33.6 MB
  int*      expert = (int*)     carve((size_t)B * 4);
  int*      slots  = (int*)     carve((size_t)3 * CAP * 4);
  _Float16* featA  = (_Float16*)carve((size_t)CAP * 64 * 1024 * 2); // 100.7 MB
  _Float16* featB  = (_Float16*)carve((size_t)CAP * 64 * 1024 * 2); // 100.7 MB

  float* out = (float*)d_out;
  const long out_main = (long)B * 3 * 128 * 128;

  dim3 blk(256);

  // 0) zero output (overflow images must read 0); counts overwritten later.
  k_zero_f32<<<4096, blk, 0, stream>>>(out, (long)out_size);

  // 1) x -> f16
  k_f32_to_f16<<<2048, blk, 0, stream>>>(x, xh, B * 3 * 1024);

  // 2) classifier: conv(3->128,k4,s4)+LReLU, 3x conv1x1(128)+LReLU, conv1x1(->32)
  k_conv_wmma<4, 4, 4, 0, 5, 3>
      <<<B, blk, conv_shmem(128, 3, 4, 4, 1024), stream>>>(
          xh, cw[0], cb[0], nullptr, 0.1f, actA, 3, 128, 128);
  k_conv_wmma<1, 1, 1, 0, 3, 128>
      <<<B, blk, conv_shmem(128, 128, 1, 1, 64), stream>>>(
          actA, cw[1], cb[1], nullptr, 0.1f, actB, 128, 128, 128);
  k_conv_wmma<1, 1, 1, 0, 3, 128>
      <<<B, blk, conv_shmem(128, 128, 1, 1, 64), stream>>>(
          actB, cw[2], cb[2], nullptr, 0.1f, actA, 128, 128, 128);
  k_conv_wmma<1, 1, 1, 0, 3, 128>
      <<<B, blk, conv_shmem(128, 128, 1, 1, 64), stream>>>(
          actA, cw[3], cb[3], nullptr, 0.1f, actB, 128, 128, 128);
  k_conv_wmma<1, 1, 1, 0, 3, 128>
      <<<B, blk, conv_shmem(32, 128, 1, 1, 64), stream>>>(
          actB, cw[4], cb[4], nullptr, 0.0f, actA, 128, 32, 32);

  // 3) pool + fc + argmax, then capacity scan -> slot tables + counts
  k_route<<<(B + 255) / 256, blk, 0, stream>>>(actA, fcw, fcb, B, 32, expert);
  k_fill_i32<<<(3 * CAP + 255) / 256, blk, 0, stream>>>(slots, 3 * CAP, -1);
  k_scan<<<1, 1, 0, stream>>>(expert, B, CAP, slots, out + out_main);

  // 4) experts (sequential, ping-pong f16 feature buffers)
  const float* prm[3][14];
  for (int e = 0; e < 3; ++e)
    for (int q = 0; q < 14; ++q)
      prm[e][q] = (const float*)d_in[13 + 14 * e + q];

  run_expert<16>(prm[0], slots + 0 * CAP, xh, featA, featB, out, stream);
  run_expert<36>(prm[1], slots + 1 * CAP, xh, featA, featB, out, stream);
  run_expert<56>(prm[2], slots + 2 * CAP, xh, featA, featB, out, stream);
}
